// HierarchicalContext_75539884802893
// MI455X (gfx1250) — matrix operand order repaired and checked by reading, not compiled, not run on hardware
//
#include <hip/hip_runtime.h>
#include <hip/hip_bf16.h>
#include <math.h>

// ---------------------------------------------------------------------------
// Types for CDNA5 WMMA (wave32): v_wmma_f32_16x16x32_bf16
// ---------------------------------------------------------------------------
typedef __bf16 bf16;
typedef __attribute__((ext_vector_type(8)))  __bf16 v8bf;
typedef __attribute__((ext_vector_type(16))) __bf16 v16bf;
typedef __attribute__((ext_vector_type(8)))  float  v8f;

#define DD   512
#define HH   8
#define CCN  3
#define DPH  64
#define TTN  128
#define BBN  64
#define SSN  128
#define HID  2048
#define BT   (BBN*TTN)        // 8192 (b*t)
#define BS   (BBN*SSN)        // 8192 (b_*s)
#define BCT  (BBN*CCN*TTN)    // 24576
#define NEGV (-1e18f)
#define KSTEP 32
#define LPAD  40              // padded LDS row stride (80B) -> conflict-free frags

__device__ __forceinline__ v8f zero8() {
  v8f z;
#pragma unroll
  for (int i = 0; i < 8; i++) z[i] = 0.0f;
  return z;
}

// A fragment (16x32, 16-bit): lane<16 -> row m=lane, K = {k0..k0+7, k0+16..k0+23}
//                             lane>=16 -> row m=lane-16, K = {k0+8..15, k0+24..31}
__device__ __forceinline__ v16bf load_a(const bf16* __restrict__ base, int ld,
                                        int row, int k0, int half) {
  const bf16* p = base + (size_t)row * ld + k0 + half * 8;
  v8bf lo = *(const v8bf*)(p);
  v8bf hi = *(const v8bf*)(p + 16);
  return __builtin_shufflevector(lo, hi, 0,1,2,3,4,5,6,7,8,9,10,11,12,13,14,15);
}

// B fragment (32x16, 16-bit), given as rows of B^T (i.e. W[N][K] row-major):
// lane<16 -> col n=lane, K = {k0..k0+7, k0+16..k0+23}; lane>=16 -> K = {k0+8..15, k0+24..31}
__device__ __forceinline__ v16bf load_b(const bf16* __restrict__ base, int ld,
                                        int row, int k0, int half) {
  const bf16* p = base + (size_t)row * ld + k0 + half * 8;
  v8bf lo = *(const v8bf*)(p);
  v8bf hi = *(const v8bf*)(p + 16);
  return __builtin_shufflevector(lo, hi, 0,1,2,3,4,5,6,7,8,9,10,11,12,13,14,15);
}

// Same fragment shapes but sourced from an LDS row (k0 folded into row pointer).
__device__ __forceinline__ v16bf lds_frag_a(const bf16* rowp, int half) {
  const bf16* p = rowp + half * 8;
  v8bf lo = *(const v8bf*)(p);
  v8bf hi = *(const v8bf*)(p + 16);
  return __builtin_shufflevector(lo, hi, 0,1,2,3,4,5,6,7,8,9,10,11,12,13,14,15);
}
__device__ __forceinline__ v16bf lds_frag_b(const bf16* rowp, int half) {
  const bf16* p = rowp + half * 8;
  v8bf lo = *(const v8bf*)(p);
  v8bf hi = *(const v8bf*)(p + 16);
  return __builtin_shufflevector(lo, hi, 0,1,2,3,4,5,6,7,8,9,10,11,12,13,14,15);
}

__device__ __forceinline__ v8f wmma_bf16(v16bf a, v16bf b, v8f c) {
  return __builtin_amdgcn_wmma_f32_16x16x32_bf16(false, a, false, b, (short)0, c,
                                                 false, false);
}

// CDNA5 async copy: 16B/lane global -> LDS, tracked by ASYNCcnt
__device__ __forceinline__ void async_cp16(unsigned lds_off, const void* gaddr) {
  asm volatile("global_load_async_to_lds_b128 %0, %1, off"
               :: "v"(lds_off), "v"(gaddr) : "memory");
}
__device__ __forceinline__ void wait_async0() {
  asm volatile("s_wait_asynccnt 0" ::: "memory");
}

// ---------------------------------------------------------------------------
// K0: doc-start indices, sentence mask, pad mask
// ---------------------------------------------------------------------------
__global__ void prep_kernel(const int* __restrict__ input_ids,
                            const int* __restrict__ ctx_index, int ndoc,
                            int* __restrict__ idx0, int* __restrict__ mask_sent,
                            int* __restrict__ pad) {
  int tid = threadIdx.x;
  if (tid < BBN) {
    int ds = 0;
    for (int i = 0; i < ndoc; i++) {
      int v = ctx_index[i];
      if (v <= tid && v > ds) ds = v;
    }
    for (int c = 0; c < CCN; c++) {
      int base = tid - CCN + c;
      int m = (base < ds) ? 1 : 0;
      mask_sent[tid * CCN + c] = m;
      idx0[tid * CCN + c] = m ? 0 : base;
    }
  }
  for (int i = tid; i < BBN * SSN; i += blockDim.x) {
    int b_i = i / SSN, s_i = i % SSN;
    pad[i] = (input_ids[s_i * BBN + b_i] == 1) ? 1 : 0;
  }
}

// ---------------------------------------------------------------------------
// Conversions
// ---------------------------------------------------------------------------
__global__ void cvt_kernel(const float* __restrict__ src, bf16* __restrict__ dst,
                           int n) {
  int i = blockIdx.x * blockDim.x + threadIdx.x;
  if (i < n) dst[i] = (bf16)src[i];
}

// context [S,B,D] -> ctx_bf [b,s,D] row-major bf16
__global__ void cvt_ctx_kernel(const float* __restrict__ context,
                               bf16* __restrict__ dst) {
  int i = blockIdx.x * blockDim.x + threadIdx.x;  // b*s*D = 2^22
  int d = i & (DD - 1);
  int r = i >> 9;
  int s_i = r & (SSN - 1);
  int b_i = r >> 7;
  dst[i] = (bf16)context[((size_t)s_i * BBN + b_i) * DD + d];
}

// Vp [b*s, D] -> Vpt [b, h, dph, s]
__global__ void vpt_kernel(const bf16* __restrict__ Vp, bf16* __restrict__ Vpt) {
  int i = blockIdx.x * blockDim.x + threadIdx.x;  // 2^22
  int s_i = i & 127;
  int dh  = (i >> 7) & 63;
  int h   = (i >> 13) & 7;
  int b_  = i >> 16;
  Vpt[i] = Vp[((size_t)b_ * SSN + s_i) * DD + h * DPH + dh];
}

// ---------------------------------------------------------------------------
// LayerNorm (ddof=1, eps added to std), optional dual gamma/beta, bf16 out.
// mode 1: source row (b,t) read from query[t*B+b]
// ---------------------------------------------------------------------------
__global__ __launch_bounds__(256)
void ln_kernel(const float* __restrict__ X, int mode,
               const float* __restrict__ g1, const float* __restrict__ bb1,
               bf16* __restrict__ o1,
               const float* __restrict__ g2, const float* __restrict__ bb2,
               bf16* __restrict__ o2) {
  int row = blockIdx.x;
  const float* src;
  if (mode == 1) {
    int b_i = row >> 7, t_i = row & 127;
    src = X + ((size_t)t_i * BBN + b_i) * DD;
  } else {
    src = X + (size_t)row * DD;
  }
  int tid = threadIdx.x;
  float x0 = src[tid], x1 = src[tid + 256];
  __shared__ float red[8];
  __shared__ float s_mu, s_rstd;
  float s = x0 + x1;
  for (int off = 16; off; off >>= 1) s += __shfl_xor(s, off, 32);
  if ((tid & 31) == 0) red[tid >> 5] = s;
  __syncthreads();
  if (tid == 0) {
    float t = 0;
    for (int i = 0; i < 8; i++) t += red[i];
    s_mu = t / (float)DD;
  }
  __syncthreads();
  float mu = s_mu;
  float d0 = x0 - mu, d1 = x1 - mu;
  float q = d0 * d0 + d1 * d1;
  for (int off = 16; off; off >>= 1) q += __shfl_xor(q, off, 32);
  if ((tid & 31) == 0) red[tid >> 5] = q;
  __syncthreads();
  if (tid == 0) {
    float t = 0;
    for (int i = 0; i < 8; i++) t += red[i];
    s_rstd = 1.0f / (sqrtf(t / (float)(DD - 1)) + 1e-6f);
  }
  __syncthreads();
  float rstd = s_rstd;
  size_t ob = (size_t)row * DD;
  o1[ob + tid]       = (bf16)(d0 * rstd * g1[tid] + bb1[tid]);
  o1[ob + tid + 256] = (bf16)(d1 * rstd * g1[tid + 256] + bb1[tid + 256]);
  if (o2) {
    o2[ob + tid]       = (bf16)(d0 * rstd * g2[tid] + bb2[tid]);
    o2[ob + tid + 256] = (bf16)(d1 * rstd * g2[tid + 256] + bb2[tid + 256]);
  }
}

// ---------------------------------------------------------------------------
// Generic WMMA GEMM with async global->LDS double-buffered staging:
//   out = (A[M,K] @ W[N,K]^T + bias)*scale [relu] [+resid]
// block = 8 waves, block tile 64x256, wave tile 32x64.
// Per K-step: A tile 64x32 (4KB) + B tile 256x32 (16KB) staged via
// GLOBAL_LOAD_ASYNC_TO_LDS_B128 (ASYNCcnt), fragments read with ds_load_b128.
// ---------------------------------------------------------------------------
__global__ __launch_bounds__(256)
void gemm_kernel(const bf16* __restrict__ Am, int lda,
                 const bf16* __restrict__ Wm, int ldw,
                 const float* __restrict__ bias, float scale, int relu,
                 const float* __restrict__ resid,
                 float* __restrict__ outF, bf16* __restrict__ outH,
                 int M, int N, int K) {
  __shared__ bf16 lA[2][64][LPAD];    // 10.0 KB
  __shared__ bf16 lB[2][256][LPAD];   // 40.0 KB
  const int tid = threadIdx.x;
  const int wave = tid >> 5;
  const int lane = tid & 31;
  const int half = lane >> 4, r = lane & 15;
  const int mblk = blockIdx.y * 64;
  const int nblk = blockIdx.x * 256;
  const int m0 = (wave >> 2) * 32;    // within block tile
  const int n0 = (wave & 3) * 64;
  const int arow  = tid >> 2;         // 0..63 (staging row)
  const int chunk = (tid & 3) * 8;    // 16B chunk offset in elements

  v8f acc[2][4];
#pragma unroll
  for (int i = 0; i < 2; i++)
#pragma unroll
    for (int j = 0; j < 4; j++) acc[i][j] = zero8();

  auto stage = [&](int k0, int buf) {
    {
      const bf16* src = Am + (size_t)(mblk + arow) * lda + k0 + chunk;
      unsigned dst = (unsigned)(uintptr_t)&lA[buf][arow][chunk];
      async_cp16(dst, src);
    }
#pragma unroll
    for (int i = 0; i < 4; i++) {
      int row = arow + i * 64;
      const bf16* src = Wm + (size_t)(nblk + row) * ldw + k0 + chunk;
      unsigned dst = (unsigned)(uintptr_t)&lB[buf][row][chunk];
      async_cp16(dst, src);
    }
  };

  stage(0, 0);
  wait_async0();
  __syncthreads();

  const int nk = K / KSTEP;
  for (int kk = 0; kk < nk; kk++) {
    const int buf = kk & 1;
    if (kk + 1 < nk) stage((kk + 1) * KSTEP, buf ^ 1);

    v16bf a0 = lds_frag_a(&lA[buf][m0 + r][0], half);
    v16bf a1 = lds_frag_a(&lA[buf][m0 + 16 + r][0], half);
    v16bf b0 = lds_frag_b(&lB[buf][n0 + r][0], half);
    v16bf b1 = lds_frag_b(&lB[buf][n0 + 16 + r][0], half);
    v16bf b2 = lds_frag_b(&lB[buf][n0 + 32 + r][0], half);
    v16bf b3 = lds_frag_b(&lB[buf][n0 + 48 + r][0], half);
    acc[0][0] = wmma_bf16(a0, b0, acc[0][0]);
    acc[0][1] = wmma_bf16(a0, b1, acc[0][1]);
    acc[0][2] = wmma_bf16(a0, b2, acc[0][2]);
    acc[0][3] = wmma_bf16(a0, b3, acc[0][3]);
    acc[1][0] = wmma_bf16(a1, b0, acc[1][0]);
    acc[1][1] = wmma_bf16(a1, b1, acc[1][1]);
    acc[1][2] = wmma_bf16(a1, b2, acc[1][2]);
    acc[1][3] = wmma_bf16(a1, b3, acc[1][3]);

    wait_async0();      // my next-stage copies landed
    __syncthreads();    // everyone's landed; everyone done reading buf
  }

#pragma unroll
  for (int i = 0; i < 2; i++)
#pragma unroll
    for (int j = 0; j < 4; j++) {
      int col = nblk + n0 + j * 16 + r;
      float bv = bias ? bias[col] : 0.0f;
#pragma unroll
      for (int e = 0; e < 8; e++) {
        int row = mblk + m0 + i * 16 + half * 8 + e;
        float v = (acc[i][j][e] + bv) * scale;
        if (relu) v = fmaxf(v, 0.0f);
        size_t o = (size_t)row * N + col;
        if (resid) v += resid[o];
        if (outF) outF[o] = v;
        if (outH) outH[o] = (bf16)v;
      }
    }
}

// ---------------------------------------------------------------------------
// Word-level scores: per (bc,h) 128x128 = Q(t x 64) * K(s x 64)^T, masked.
// One 16x16 tile per wave, K=64. Written f32 straight into attn_word output.
// ---------------------------------------------------------------------------
__global__ __launch_bounds__(256)
void scores_kernel(const bf16* __restrict__ Qp, const bf16* __restrict__ Kp,
                   const int* __restrict__ idx0, const int* __restrict__ mask_sent,
                   const int* __restrict__ pad, float* __restrict__ scores) {
  int gtile = blockIdx.x * 8 + (threadIdx.x >> 5);
  int lane = threadIdx.x & 31;
  int half = lane >> 4, r = lane & 15;
  int tn = gtile & 7;
  int tm = (gtile >> 3) & 7;
  int h  = (gtile >> 6) & 7;
  int bc = gtile >> 9;               // 0..191
  int sent = idx0[bc];
  int ms = mask_sent[bc];
  int b_i = bc / CCN;
  const bf16* Ab = Qp + (size_t)(b_i * TTN) * DD + h * DPH;
  const bf16* Bb = Kp + (size_t)(sent * SSN) * DD + h * DPH;
  v8f acc = zero8();
#pragma unroll
  for (int ks = 0; ks < 64; ks += 32) {
    v16bf a = load_a(Ab, DD, tm * 16 + r, ks, half);
    v16bf b = load_b(Bb, DD, tn * 16 + r, ks, half);
    acc = wmma_bf16(a, b, acc);
  }
  size_t base = (size_t)(bc * HH + h) * TTN * SSN;
  int col = tn * 16 + r;
  int masked = ms | pad[sent * SSN + col];
#pragma unroll
  for (int e = 0; e < 8; e++) {
    int row = tm * 16 + half * 8 + e;
    scores[base + (size_t)row * SSN + col] = masked ? NEGV : acc[e];
  }
}

// Softmax over s=128, one wave per row; in-place f32 + bf16 copy.
__global__ __launch_bounds__(256)
void softmax_kernel(float* __restrict__ attn, bf16* __restrict__ attn_bf) {
  int row = blockIdx.x * 8 + (threadIdx.x >> 5);
  int lane = threadIdx.x & 31;
  float* p = attn + (size_t)row * SSN;
  float v[4];
  float mx = -INFINITY;
#pragma unroll
  for (int i = 0; i < 4; i++) {
    v[i] = p[lane + i * 32];
    mx = fmaxf(mx, v[i]);
  }
  for (int off = 16; off; off >>= 1) mx = fmaxf(mx, __shfl_xor(mx, off, 32));
  float sum = 0.0f;
#pragma unroll
  for (int i = 0; i < 4; i++) {
    v[i] = expf(v[i] - mx);
    sum += v[i];
  }
  for (int off = 16; off; off >>= 1) sum += __shfl_xor(sum, off, 32);
  float inv = 1.0f / sum;
#pragma unroll
  for (int i = 0; i < 4; i++) {
    float a = v[i] * inv;
    p[lane + i * 32] = a;
    attn_bf[(size_t)row * SSN + lane + i * 32] = (bf16)a;
  }
}

// attn @ V: per (bc,h): (t x s) * (s x 64); B^T rows come from Vpt [b,h,dph,s]
__global__ __launch_bounds__(256)
void attnv_kernel(const bf16* __restrict__ attn_bf, const bf16* __restrict__ Vpt,
                  const int* __restrict__ idx0, bf16* __restrict__ ctxw) {
  int blk = blockIdx.x;       // bc*8 + h
  int h = blk & 7, bc = blk >> 3;
  int sent = idx0[bc];
  int wave = threadIdx.x >> 5;
  int lane = threadIdx.x & 31;
  int half = lane >> 4, r = lane & 15;
  const bf16* Ab = attn_bf + (size_t)(bc * HH + h) * TTN * SSN;        // lda=128
  const bf16* Bb = Vpt + (size_t)(sent * HH + h) * DPH * SSN;          // ldb=128
  v8f acc[4];
#pragma unroll
  for (int j = 0; j < 4; j++) acc[j] = zero8();
  for (int k0 = 0; k0 < SSN; k0 += 32) {
    v16bf a = load_a(Ab, SSN, wave * 16 + r, k0, half);
#pragma unroll
    for (int j = 0; j < 4; j++) {
      v16bf b = load_b(Bb, SSN, j * 16 + r, k0, half);
      acc[j] = wmma_bf16(a, b, acc[j]);
    }
  }
#pragma unroll
  for (int j = 0; j < 4; j++) {
    int col = j * 16 + r;
#pragma unroll
    for (int e = 0; e < 8; e++) {
      int row = wave * 16 + half * 8 + e;
      ctxw[((size_t)bc * TTN + row) * DD + h * DPH + col] = (bf16)acc[j][e];
    }
  }
}

// ---------------------------------------------------------------------------
// Sentence-level attention (K over C=3), one wave per (b,t,head)
// ---------------------------------------------------------------------------
__global__ __launch_bounds__(256)
void sent_attn_kernel(const bf16* __restrict__ Qs, const bf16* __restrict__ Ks,
                      const bf16* __restrict__ Vs, const int* __restrict__ mask_sent,
                      float* __restrict__ attn_sent, bf16* __restrict__ ctxs) {
  int row = blockIdx.x;               // b*t
  int b_i = row >> 7, t_i = row & 127;
  int h = threadIdx.x >> 5, lane = threadIdx.x & 31;
  const bf16* q = Qs + (size_t)row * DD + h * DPH;
  float q0 = (float)q[lane * 2], q1 = (float)q[lane * 2 + 1];
  float sc3[CCN];
  int krow[CCN];
#pragma unroll
  for (int c = 0; c < CCN; c++) {
    krow[c] = (b_i * CCN + c) * TTN + t_i;
    const bf16* kp = Ks + (size_t)krow[c] * DD + h * DPH;
    float pr = q0 * (float)kp[lane * 2] + q1 * (float)kp[lane * 2 + 1];
    for (int off = 16; off; off >>= 1) pr += __shfl_xor(pr, off, 32);
    sc3[c] = mask_sent[b_i * CCN + c] ? NEGV : pr;
  }
  float m = fmaxf(sc3[0], fmaxf(sc3[1], sc3[2]));
  float e0 = expf(sc3[0] - m), e1 = expf(sc3[1] - m), e2 = expf(sc3[2] - m);
  float inv = 1.0f / (e0 + e1 + e2);
  float a0 = e0 * inv, a1 = e1 * inv, a2 = e2 * inv;
  if (lane < CCN) {
    float av = (lane == 0) ? a0 : ((lane == 1) ? a1 : a2);
    attn_sent[((size_t)row * HH + h) * CCN + lane] = av;
  }
  float o0 = 0.0f, o1 = 0.0f;
#pragma unroll
  for (int c = 0; c < CCN; c++) {
    float ac = (c == 0) ? a0 : ((c == 1) ? a1 : a2);
    const bf16* vp = Vs + (size_t)krow[c] * DD + h * DPH;
    o0 += ac * (float)vp[lane * 2];
    o1 += ac * (float)vp[lane * 2 + 1];
  }
  ctxs[(size_t)row * DD + h * DPH + lane * 2]     = (bf16)o0;
  ctxs[(size_t)row * DD + h * DPH + lane * 2 + 1] = (bf16)o1;
}

// concat([q, sc]) -> bf16 [b*t, 1024]
__global__ void concat_kernel(const float* __restrict__ query,
                              const float* __restrict__ sc, bf16* __restrict__ cat) {
  int i = blockIdx.x * blockDim.x + threadIdx.x;  // BT*1024
  int d = i & 1023;
  int row = i >> 10;
  int b_i = row >> 7, t_i = row & 127;
  float v;
  if (d < DD) v = query[((size_t)t_i * BBN + b_i) * DD + d];
  else        v = sc[(size_t)row * DD + (d - DD)];
  cat[i] = (bf16)v;
}

// out = LN_s((1-g)*q + g*sc), g=sigmoid(logits); output layout [T,B,D]
__global__ __launch_bounds__(256)
void final_kernel(const float* __restrict__ query, const float* __restrict__ logits,
                  const float* __restrict__ sc, const float* __restrict__ g,
                  const float* __restrict__ bb, float* __restrict__ out) {
  int row = blockIdx.x;
  int b_i = row >> 7, t_i = row & 127;
  const float* qrow = query + ((size_t)t_i * BBN + b_i) * DD;
  int tid = threadIdx.x;
  float y0, y1;
  {
    float gt = 1.0f / (1.0f + expf(-logits[(size_t)row * DD + tid]));
    y0 = (1.0f - gt) * qrow[tid] + gt * sc[(size_t)row * DD + tid];
  }
  {
    int d = tid + 256;
    float gt = 1.0f / (1.0f + expf(-logits[(size_t)row * DD + d]));
    y1 = (1.0f - gt) * qrow[d] + gt * sc[(size_t)row * DD + d];
  }
  __shared__ float red[8];
  __shared__ float s_mu, s_rstd;
  float s = y0 + y1;
  for (int off = 16; off; off >>= 1) s += __shfl_xor(s, off, 32);
  if ((tid & 31) == 0) red[tid >> 5] = s;
  __syncthreads();
  if (tid == 0) {
    float t = 0;
    for (int i = 0; i < 8; i++) t += red[i];
    s_mu = t / (float)DD;
  }
  __syncthreads();
  float mu = s_mu;
  float d0 = y0 - mu, d1 = y1 - mu;
  float q2 = d0 * d0 + d1 * d1;
  for (int off = 16; off; off >>= 1) q2 += __shfl_xor(q2, off, 32);
  if ((tid & 31) == 0) red[tid >> 5] = q2;
  __syncthreads();
  if (tid == 0) {
    float t = 0;
    for (int i = 0; i < 8; i++) t += red[i];
    s_rstd = 1.0f / (sqrtf(t / (float)(DD - 1)) + 1e-6f);
  }
  __syncthreads();
  float rstd = s_rstd;
  size_t ob = ((size_t)t_i * BBN + b_i) * DD;
  out[ob + tid]       = d0 * rstd * g[tid] + bb[tid];
  out[ob + tid + 256] = d1 * rstd * g[tid + 256] + bb[tid + 256];
}

// ---------------------------------------------------------------------------
// Host orchestration
// ---------------------------------------------------------------------------
static void launch_gemm(hipStream_t stream, const bf16* Am, int lda, const bf16* Wm,
                        int ldw, const float* bias, float scale, int relu,
                        const float* resid, float* outF, bf16* outH, int M, int N,
                        int K) {
  dim3 grid(N / 256, M / 64);
  gemm_kernel<<<grid, 256, 0, stream>>>(Am, lda, Wm, ldw, bias, scale, relu, resid,
                                        outF, outH, M, N, K);
}

extern "C" void kernel_launch(void* const* d_in, const int* in_sizes, int n_in,
                              void* d_out, int out_size, void* d_ws, size_t ws_size,
                              hipStream_t stream) {
  const int*   input_ids = (const int*)d_in[0];
  const float* query     = (const float*)d_in[1];
  const float* context   = (const float*)d_in[2];
  const int*   ctx_index = (const int*)d_in[3];
  const int    ndoc      = in_sizes[3];
  const float* wkW = (const float*)d_in[4];   const float* wkB = (const float*)d_in[5];
  const float* wvW = (const float*)d_in[6];   const float* wvB = (const float*)d_in[7];
  const float* wqW = (const float*)d_in[8];   const float* wqB = (const float*)d_in[9];
  const float* woW = (const float*)d_in[10];  const float* woB = (const float*)d_in[11];
  const float* skW = (const float*)d_in[12];  const float* skB = (const float*)d_in[13];
  const float* svW = (const float*)d_in[14];  const float* svB = (const float*)d_in[15];
  const float* sqW = (const float*)d_in[16];  const float* sqB = (const float*)d_in[17];
  const float* soW = (const float*)d_in[18];  const float* soB = (const float*)d_in[19];
  const float* ffW1 = (const float*)d_in[20]; const float* ffb1 = (const float*)d_in[21];
  const float* ffW2 = (const float*)d_in[22]; const float* ffb2 = (const float*)d_in[23];
  const float* gW = (const float*)d_in[24];   const float* gB = (const float*)d_in[25];
  const float* ln_qw_g = (const float*)d_in[26]; const float* ln_qw_b = (const float*)d_in[27];
  const float* ln_qs_g = (const float*)d_in[28]; const float* ln_qs_b = (const float*)d_in[29];
  const float* ln_w_g  = (const float*)d_in[30]; const float* ln_w_b  = (const float*)d_in[31];
  const float* ln_s_g  = (const float*)d_in[32]; const float* ln_s_b  = (const float*)d_in[33];
  const float* ln_ff_g = (const float*)d_in[34]; const float* ln_ff_b = (const float*)d_in[35];

  float* out = (float*)d_out;
  float* attn_word = out + (size_t)TTN * BBN * DD;                    // 4,194,304
  float* attn_sent = attn_word + (size_t)BBN * CCN * HH * TTN * SSN;  // +25,165,824

  char* ws = (char*)d_ws;
  size_t off = 0;
  auto alloc = [&](size_t bytes) {
    size_t r = off;
    off += (bytes + 255) & ~(size_t)255;
    return r;
  };
  int* idx0      = (int*)(ws + alloc(BBN * CCN * 4));
  int* mask_sent = (int*)(ws + alloc(BBN * CCN * 4));
  int* pad       = (int*)(ws + alloc(BBN * SSN * 4));
  bf16* wkWb = (bf16*)(ws + alloc((size_t)DD * DD * 2));
  bf16* wvWb = (bf16*)(ws + alloc((size_t)DD * DD * 2));
  bf16* wqWb = (bf16*)(ws + alloc((size_t)DD * DD * 2));
  bf16* woWb = (bf16*)(ws + alloc((size_t)DD * DD * 2));
  bf16* skWb = (bf16*)(ws + alloc((size_t)DD * DD * 2));
  bf16* svWb = (bf16*)(ws + alloc((size_t)DD * DD * 2));
  bf16* sqWb = (bf16*)(ws + alloc((size_t)DD * DD * 2));
  bf16* soWb = (bf16*)(ws + alloc((size_t)DD * DD * 2));
  bf16* ffW1b = (bf16*)(ws + alloc((size_t)HID * DD * 2));
  bf16* ffW2b = (bf16*)(ws + alloc((size_t)DD * HID * 2));
  bf16* gWb   = (bf16*)(ws + alloc((size_t)DD * 2 * DD * 2));
  bf16* qword = (bf16*)(ws + alloc((size_t)BT * DD * 2));
  bf16* qsent = (bf16*)(ws + alloc((size_t)BT * DD * 2));
  bf16* ctxbf = (bf16*)(ws + alloc((size_t)BS * DD * 2));
  bf16* Kp    = (bf16*)(ws + alloc((size_t)BS * DD * 2));
  bf16* Vp    = (bf16*)(ws + alloc((size_t)BS * DD * 2));
  bf16* Vpt   = (bf16*)(ws + alloc((size_t)BS * DD * 2));
  bf16* Qp    = (bf16*)(ws + alloc((size_t)BT * DD * 2));
  bf16* attnbf = (bf16*)(ws + alloc((size_t)BBN * CCN * HH * TTN * SSN * 2)); // 50.3MB
  bf16* ctxw  = (bf16*)(ws + alloc((size_t)BCT * DD * 2));                    // 25.2MB
  float* cw   = (float*)(ws + alloc((size_t)BCT * DD * 4));                   // 50.3MB
  bf16* cs    = (bf16*)(ws + alloc((size_t)BCT * DD * 2));
  bf16* Ks    = (bf16*)(ws + alloc((size_t)BCT * DD * 2));
  bf16* Vs    = (bf16*)(ws + alloc((size_t)BCT * DD * 2));
  bf16* Qs    = (bf16*)(ws + alloc((size_t)BT * DD * 2));
  bf16* ctxs  = (bf16*)(ws + alloc((size_t)BT * DD * 2));
  // Aliases over dead regions (all 256B-aligned, strictly later lifetimes):
  bf16*  hmid   = (bf16*)attnbf;   // 32MB <= 50.3MB, dead after attnv
  float* sc0    = (float*)ctxw;    // 16MB <= 25.2MB, dead after word o-proj
  float* sc     = (float*)cw;      // 16MB <= 50.3MB, dead after ln_w
  bf16*  ffin   = (bf16*)qword;    // dead after Qp projection
  bf16*  cat    = (bf16*)Ks;       // dead after sent_attn
  float* logits = (float*)Vs;      // dead after sent_attn

  // 0) masks / indices
  prep_kernel<<<1, 256, 0, stream>>>(input_ids, ctx_index, ndoc, idx0, mask_sent, pad);

  // 1) weights -> bf16
  struct { const float* s; bf16* d; int n; } jobs[11] = {
      {wkW, wkWb, DD * DD},   {wvW, wvWb, DD * DD},   {wqW, wqWb, DD * DD},
      {woW, woWb, DD * DD},   {skW, skWb, DD * DD},   {svW, svWb, DD * DD},
      {sqW, sqWb, DD * DD},   {soW, soWb, DD * DD},   {ffW1, ffW1b, HID * DD},
      {ffW2, ffW2b, DD * HID}, {gW, gWb, DD * 2 * DD}};
  for (int i = 0; i < 11; i++)
    cvt_kernel<<<(jobs[i].n + 255) / 256, 256, 0, stream>>>(jobs[i].s, jobs[i].d,
                                                            jobs[i].n);

  // 2) LN(q) -> q_word, q_sent (dual)
  ln_kernel<<<BT, 256, 0, stream>>>(query, 1, ln_qw_g, ln_qw_b, qword, ln_qs_g,
                                    ln_qs_b, qsent);
  // 3) context -> bf16 [b,s,D]
  cvt_ctx_kernel<<<(BS * DD) / 256, 256, 0, stream>>>(context, ctxbf);

  // 4) word projections (k/v once per sentence, not per gathered copy)
  launch_gemm(stream, ctxbf, DD, wkWb, DD, wkB, 1.0f, 0, nullptr, nullptr, Kp, BS, DD, DD);
  launch_gemm(stream, ctxbf, DD, wvWb, DD, wvB, 1.0f, 0, nullptr, nullptr, Vp, BS, DD, DD);
  vpt_kernel<<<(BS * DD) / 256, 256, 0, stream>>>(Vp, Vpt);
  launch_gemm(stream, qword, DD, wqWb, DD, wqB, 0.125f, 0, nullptr, nullptr, Qp, BT, DD, DD);

  // 5) scores -> attn_word slice of d_out, softmax in place + bf16 copy
  scores_kernel<<<(BBN * CCN * HH * 64) / 8, 256, 0, stream>>>(Qp, Kp, idx0, mask_sent,
                                                               pad, attn_word);
  softmax_kernel<<<(BBN * CCN * HH * TTN) / 8, 256, 0, stream>>>(attn_word, attnbf);

  // 6) attn @ V -> ctxw, then o-projection -> cw (f32)
  attnv_kernel<<<BBN * CCN * HH, 256, 0, stream>>>(attnbf, Vpt, idx0, ctxw);
  launch_gemm(stream, ctxw, DD, woWb, DD, woB, 1.0f, 0, nullptr, cw, nullptr, BCT, DD, DD);

  // 7) cs = LN_w(cw)
  ln_kernel<<<BCT, 256, 0, stream>>>(cw, 0, ln_w_g, ln_w_b, cs, nullptr, nullptr,
                                     nullptr);

  // 8) sentence projections
  launch_gemm(stream, cs, DD, skWb, DD, skB, 1.0f, 0, nullptr, nullptr, Ks, BCT, DD, DD);
  launch_gemm(stream, cs, DD, svWb, DD, svB, 1.0f, 0, nullptr, nullptr, Vs, BCT, DD, DD);
  launch_gemm(stream, qsent, DD, sqWb, DD, sqB, 0.125f, 0, nullptr, nullptr, Qs, BT, DD, DD);

  // 9) sentence attention (+ attn_sent output), then o-projection -> sc0
  sent_attn_kernel<<<BT, 256, 0, stream>>>(Qs, Ks, Vs, mask_sent, attn_sent, ctxs);
  launch_gemm(stream, ctxs, DD, soWb, DD, soB, 1.0f, 0, nullptr, sc0, nullptr, BT, DD, DD);

  // 10) FFN: sc = W2(relu(W1(LN_ff(sc0)))) + sc0
  ln_kernel<<<BT, 256, 0, stream>>>(sc0, 0, ln_ff_g, ln_ff_b, ffin, nullptr, nullptr,
                                    nullptr);
  launch_gemm(stream, ffin, DD, ffW1b, DD, ffb1, 1.0f, 1, nullptr, nullptr, hmid, BT,
              HID, DD);
  launch_gemm(stream, hmid, HID, ffW2b, HID, ffb2, 1.0f, 0, sc0, sc, nullptr, BT, DD,
              HID);

  // 11) gate + final LN -> out [T,B,D]
  concat_kernel<<<(BT * 2 * DD) / 256, 256, 0, stream>>>(query, sc, cat);
  launch_gemm(stream, cat, 2 * DD, gWb, 2 * DD, gB, 1.0f, 0, nullptr, logits, nullptr,
              BT, DD, 2 * DD);
  final_kernel<<<BT, 256, 0, stream>>>(query, logits, sc, ln_s_g, ln_s_b, out);

  (void)n_in; (void)out_size; (void)ws_size;
}
